// EMRouting_41463614275717
// MI455X (gfx1250) — matrix operand order repaired and checked by reading, not compiled, not run on hardware
//
#include <hip/hip_runtime.h>

// ---------------------------------------------------------------------------
// EM capsule routing, MI455X (gfx1250, wave32).
// One 512-thread workgroup per (b,h,w) position. Whole vote slab staged once
// from HBM into LDS as bf16 ([o][n][p], 288 KB); all EM phases run from LDS.
// Heavy n=288 reductions and the E-step pose dot-products use
// v_wmma_f32_16x16x32_bf16. Activations are fetched with
// global_load_async_to_lds_b32 (ASYNCcnt path).
// ---------------------------------------------------------------------------

typedef __attribute__((ext_vector_type(16))) __bf16 v16bf;
typedef __attribute__((ext_vector_type(8)))  __bf16 v8bf;
typedef __attribute__((ext_vector_type(4)))  __bf16 v4bf;
typedef __attribute__((ext_vector_type(8)))  float  v8f;

#define NPOS 576        // B*H*W = 16*6*6
#define NK   288        // KH*KW*I = 3*3*32
#define NO   32         // output capsules
#define NP   16         // pose entries (4x4)
#define SLAB (NK*NO*NP) // 147456 f32 per position
#define EPSC 1e-7f

// LDS layout (bytes):
//   [0,294912)        bf16 votes   sV[o][n][p]
//   [294912,313344)   bf16 scratch sLN[n][o]   (log_num, then R_final)
//   [313344,...)      f32 stats region
#define LDS_BYTES 326592

__global__ __launch_bounds__(512, 1)
void em_routing_gfx1250(const float* __restrict__ votes,
                        const float* __restrict__ acts,
                        const float* __restrict__ beta_a,
                        const float* __restrict__ beta_u,
                        float* __restrict__ out)
{
    extern __shared__ char smem[];
    __bf16* sV  = (__bf16*)smem;                         // [NO][NK][NP]
    __bf16* sLN = (__bf16*)(smem + SLAB * 2);            // [NK][NO]
    float*  f32 = (float*)(smem + SLAB * 2 + NK * NO * 2);
    float* sS1   = f32;           // 512 : Sum R*V   per (o,p)
    float* sS2   = sS1 + 512;     // 512 : Sum R*V^2 per (o,p) (reused: sigma)
    float* sMu   = sS2 + 512;     // 512
    float* sW    = sMu + 512;     // 512 : 1/(2 sigma^2)
    float* sU    = sW  + 512;     // 512 : mu/sigma^2
    float* sA    = sU  + 512;     // 288 : activations
    float* sR1   = sA  + 288;     // 288 : M-step-1 assignments (o-independent)
    float* sK1   = sR1 + 288;     // 32  : per-o log_p constant
    float* sLogA = sK1 + 32;      // 32  : log(a_j + eps) after M-step 1
    float* sLD   = sLogA + 32;    // 32  : logsumexp denominator per i
    float* sSumRf= sLD + 32;      // 32  : sum of final R per o
    float* sMisc = sSumRf + 32;   // 16

    const int tid  = threadIdx.x;
    const int lane = tid & 31;
    const int wv   = tid >> 5;          // wave id 0..15
    const bool hi  = lane >= 16;
    const int nl   = lane & 15;
    const int pos  = blockIdx.x;

    const float* vbase = votes + (size_t)pos * SLAB;
    const float* abase = acts  + (size_t)pos * NK;

    // -------- async DMA: activations -> LDS (ASYNCcnt) --------
    if (tid < NK) {
        unsigned ldsAddr = (unsigned)(unsigned long long)(&sA[tid]);
        unsigned voff    = (unsigned)(tid * 4);
        unsigned long long base = (unsigned long long)abase;
        asm volatile("global_load_async_to_lds_b32 %0, %1, %2"
                     :: "v"(ldsAddr), "v"(voff), "s"(base) : "memory");
    }

    // -------- stage votes: f32 global (b128) -> bf16 LDS [o][n][p] --------
    const float4* v4 = (const float4*)vbase;
    for (int c = 0; c < 72; ++c) {
        int f4 = c * 512 + tid;
        if (c < 71) __builtin_prefetch(v4 + f4 + 512, 0, 0);
        float4 x = v4[f4];
        int f = f4 << 2;                    // = n*512 + o*16 + p
        int p = f & 15, o = (f >> 4) & 31, n = f >> 9;
        v4bf w;
        w[0] = (__bf16)x.x; w[1] = (__bf16)x.y;
        w[2] = (__bf16)x.z; w[3] = (__bf16)x.w;
        *(v4bf*)&sV[(o * NK + n) * NP + p] = w;   // ds_store_b64
    }
    asm volatile("s_wait_asynccnt 0" ::: "memory");
    __syncthreads();

    // -------- M-step 1 assignments: R1[n] = (0.8*a_i + 0.2)/32 --------
    if (tid < NK) sR1[tid] = (0.8f * sA[tid] + 0.2f) * (1.0f / 32.0f);
    __syncthreads();
    if (tid == 0) {
        float s = 0.f;
        for (int n = 0; n < NK; ++n) s += sR1[n];
        sMisc[0] = s;                       // sum_R (same for every o)
    }
    __syncthreads();

    // A-fragment K map (16-bit A matrix 16x32, ISA 7.12.2)
    int kA[16];
    #pragma unroll
    for (int e = 0; e < 16; ++e) {
        int j = e >> 1, t = e & 1, b = hi ? 8 : 0;
        kA[e] = (j < 4) ? (2 * j + t + b) : (16 + 2 * (j - 4) + t + b);
    }

    // -------- Phase A: WMMA reductions  Sum R1*V, Sum R1*V^2 --------
    #pragma unroll
    for (int oo = 0; oo < 2; ++oo) {
        const int o = wv + oo * 16;
        v8f acc1 = {0.f,0.f,0.f,0.f,0.f,0.f,0.f,0.f};
        v8f acc2 = {0.f,0.f,0.f,0.f,0.f,0.f,0.f,0.f};
        for (int cch = 0; cch < 9; ++cch) {
            const int n0 = cch * 32;
            v16bf afrag, bfrag, bsq;
            #pragma unroll
            for (int e = 0; e < 16; ++e)
                afrag[e] = (__bf16)sR1[n0 + kA[e]];          // A[m,k]=R1[n0+k]
            #pragma unroll
            for (int e = 0; e < 16; ++e) {
                __bf16 vv = sV[(o * NK + n0 + kA[e]) * NP + nl];
                bfrag[e] = vv;
                float vf = (float)vv;
                bsq[e] = (__bf16)(vf * vf);
            }
            acc1 = __builtin_amdgcn_wmma_f32_16x16x32_bf16(false, afrag, false, bfrag, (short)0, acc1, false, false);
            acc2 = __builtin_amdgcn_wmma_f32_16x16x32_bf16(false, afrag, false, bsq,   (short)0, acc2, false, false);
        }
        if (!hi) {                       // D row M=0: VGPR0, N = lane
            sS1[o * NP + nl] = acc1[0];
            sS2[o * NP + nl] = acc2[0];
        }
    }
    __syncthreads();

    // -------- M-step 1 statistics --------
    {
        const float sumR = sMisc[0];
        float S1 = sS1[tid], S2 = sS2[tid];
        float denom = sumR + EPSC;
        float mu  = S1 / denom;
        float sig = (S2 - 2.f * mu * S1 + mu * mu * sumR) / denom;
        sMu[tid] = mu;
        sW[tid]  = 0.5f / sig;
        sU[tid]  = mu / sig;
        sS2[tid] = sig;                 // keep sigma for serial per-o loop
    }
    __syncthreads();
    if (tid < NO) {
        const int o = tid;
        const float sumR = sMisc[0];
        float bu = beta_u[o];
        float k1 = 0.f, cost = 0.f;
        for (int p = 0; p < NP; ++p) {
            float sig = sS2[o * NP + p];
            float mu  = sMu[o * NP + p];
            float w   = sW[o * NP + p];
            k1   -= __logf(6.2831853f * sig + EPSC) + mu * mu * w;
            cost += (bu - 0.5f * __logf(sig + EPSC)) * sumR;
        }
        sK1[o] = k1;
        const float it1 = 0.01f * (1.f - 0.95f);
        float aj = 1.f / (1.f + __expf(-it1 * (beta_a[o] - cost)));
        sLogA[o] = __logf(aj + EPSC);
    }
    __syncthreads();

    // -------- Phase B1: E-step log_num via WMMA (A=[V|V^2], B=[u|-w]) -----
    #pragma unroll
    for (int oo = 0; oo < 2; ++oo) {
        const int o = wv + oo * 16;
        v16bf buw;
        #pragma unroll
        for (int e = 0; e < 16; ++e) {
            int K = kA[e];
            float val = (K < 16) ? sU[o * NP + K] : -sW[o * NP + (K - 16)];
            buw[e] = (__bf16)val;
        }
        const float lc = sLogA[o] + sK1[o];
        for (int cch = 0; cch < 18; ++cch) {
            const int n = cch * 16 + nl;
            v8bf ch = *(const v8bf*)&sV[(o * NK + n) * NP + (hi ? 8 : 0)];
            v16bf afrag;
            #pragma unroll
            for (int e = 0; e < 16; ++e) {
                if (e < 8) afrag[e] = ch[e];
                else { float vf = (float)ch[e - 8]; afrag[e] = (__bf16)(vf * vf); }
            }
            v8f d = {0.f,0.f,0.f,0.f,0.f,0.f,0.f,0.f};
            d = __builtin_amdgcn_wmma_f32_16x16x32_bf16(false, afrag, false, buw, (short)0, d, false, false);
            if (nl == 0) {
                int nb = cch * 16 + (hi ? 8 : 0);
                #pragma unroll
                for (int r = 0; r < 8; ++r)
                    sLN[(nb + r) * NO + o] = (__bf16)(d[r] + lc);
            }
        }
    }
    __syncthreads();

    // -------- logsumexp over (kh,kw,o) per input capsule i --------
    if (tid < NO) {
        const int i = tid;
        float m = -3.0e38f;
        for (int kk = 0; kk < 9; ++kk) {
            const __bf16* row = &sLN[(kk * NO + i) * NO];
            for (int o = 0; o < NO; ++o) m = fmaxf(m, (float)row[o]);
        }
        float s = 0.f;
        for (int kk = 0; kk < 9; ++kk) {
            const __bf16* row = &sLN[(kk * NO + i) * NO];
            for (int o = 0; o < NO; ++o) s += __expf((float)row[o] - m);
        }
        sLD[i] = m + __logf(s);
    }
    __syncthreads();

    // -------- final R in place: R_f = 0.8*a_i*exp(ln - ld) + 0.2*R1 -------
    for (int q = tid; q < NK * NO; q += 512) {
        int n = q >> 5, o = q & 31, i = n & 31; (void)o;
        float r2 = __expf((float)sLN[q] - sLD[i]);
        float rf = 0.8f * sA[n] * r2 + 0.2f * sR1[n];
        sLN[q] = (__bf16)rf;
    }
    __syncthreads();

    // -------- Phase B2: WMMA reductions with A = R_f (o-dependent) --------
    #pragma unroll
    for (int oo = 0; oo < 2; ++oo) {
        const int o = wv + oo * 16;
        v8f acc1 = {0.f,0.f,0.f,0.f,0.f,0.f,0.f,0.f};
        v8f acc2 = {0.f,0.f,0.f,0.f,0.f,0.f,0.f,0.f};
        for (int cch = 0; cch < 9; ++cch) {
            const int n0 = cch * 32;
            v16bf afrag, bfrag, bsq;
            #pragma unroll
            for (int e = 0; e < 16; ++e)
                afrag[e] = sLN[(n0 + kA[e]) * NO + o];
            #pragma unroll
            for (int e = 0; e < 16; ++e) {
                __bf16 vv = sV[(o * NK + n0 + kA[e]) * NP + nl];
                bfrag[e] = vv;
                float vf = (float)vv;
                bsq[e] = (__bf16)(vf * vf);
            }
            acc1 = __builtin_amdgcn_wmma_f32_16x16x32_bf16(false, afrag, false, bfrag, (short)0, acc1, false, false);
            acc2 = __builtin_amdgcn_wmma_f32_16x16x32_bf16(false, afrag, false, bsq,   (short)0, acc2, false, false);
        }
        if (!hi) {
            sS1[o * NP + nl] = acc1[0];
            sS2[o * NP + nl] = acc2[0];
        }
    }
    if (tid < NO) {                       // deterministic sum of final R per o
        float s = 0.f;
        for (int n = 0; n < NK; ++n) s += (float)sLN[n * NO + tid];
        sSumRf[tid] = s;
    }
    __syncthreads();

    // -------- final statistics + outputs --------
    {
        const int o = tid >> 4;
        float Srf = sSumRf[o];
        float denom = Srf + EPSC;
        float S1 = sS1[tid], S2 = sS2[tid];
        float mu  = S1 / denom;
        float sig = (S2 - 2.f * mu * S1 + mu * mu * Srf) / denom;
        out[(size_t)pos * 512 + tid] = mu;           // poses (b,h,w,o,4,4)
        sS2[tid] = sig;
    }
    __syncthreads();
    if (tid < NO) {
        const int o = tid;
        float Srf = sSumRf[o];
        float bu = beta_u[o];
        float cost = 0.f;
        for (int p = 0; p < NP; ++p)
            cost += (bu - 0.5f * __logf(sS2[o * NP + p] + EPSC)) * Srf;
        const float it2 = 0.01f * (1.f - 0.857375f); // 1 - 0.95^3
        float aj = 1.f / (1.f + __expf(-it2 * (beta_a[o] - cost)));
        out[(size_t)NPOS * 512 + (size_t)pos * NO + o] = aj;   // acts
    }
}

extern "C" void kernel_launch(void* const* d_in, const int* in_sizes, int n_in,
                              void* d_out, int out_size, void* d_ws, size_t ws_size,
                              hipStream_t stream)
{
    (void)in_sizes; (void)n_in; (void)out_size; (void)d_ws; (void)ws_size;
    const float* votes  = (const float*)d_in[0];
    const float* acts   = (const float*)d_in[1];
    const float* beta_a = (const float*)d_in[2];
    const float* beta_u = (const float*)d_in[3];
    float* out = (float*)d_out;
    em_routing_gfx1250<<<dim3(NPOS), dim3(512), LDS_BYTES, stream>>>(
        votes, acts, beta_a, beta_u, out);
}
